// DoubleTrain_53223234732118
// MI455X (gfx1250) — matrix operand order repaired and checked by reading, not compiled, not run on hardware
//
#include <hip/hip_runtime.h>
#include <hip/hip_bf16.h>
#include <cstdint>
#include <cstddef>

// ---------------------------------------------------------------------------
// DoubleTrain (GRU + shared-table RNN) for MI455X / gfx1250, wave32 + WMMA.
// B=4096, T=200, F=64, UNITS=48, G=144, SHARED=15, ID_COL=2.
// ---------------------------------------------------------------------------

#define B_      4096
#define T_      200
#define F_      64
#define U_      48
#define G_      144
#define SHARED_ 15

typedef __attribute__((ext_vector_type(16))) _Float16 v16h;
typedef __attribute__((ext_vector_type(8)))  float    v8f;

// Hardware transcendental path: v_exp_f32 + v_rcp_f32 (no IEEE divide).
__device__ __forceinline__ float sigf(float x) {
    return __builtin_amdgcn_rcpf(1.0f + __expf(-x));
}
// tanh(x) = 1 - 2/(exp(2x)+1): saturates correctly at +/-1, no division.
__device__ __forceinline__ float tanh_fast(float x) {
    return 1.0f - 2.0f * __builtin_amdgcn_rcpf(1.0f + __expf(2.0f * x));
}

// A-fragment inverse map: within a 16x32 f16 A tile, lane L = M%16 + 16*hi,
// element e of v16h holds K = e + 8*(e>=8) + 8*hi.  Inverse for Kr = K%32:
//   hi = bit3 of Kr,  e = (Kr&7) + 8*(Kr>=16)
__device__ __forceinline__ void amap(int Kr, int& hi, int& e) {
    hi = (Kr >> 3) & 1;
    e  = (Kr & 7) + ((Kr >> 4) << 3);
}

// ---------------------------------------------------------------------------
// Kernel 1: fused card-GRU.  grid = B/64 workgroups, 128 threads (4 waves).
// Wave w owns batch rows [64*blk + 16*w, +16).  Hidden state h lives in v8f
// registers in WMMA C-layout; per step:
//   h_f16 -> LDS (A frag layout), x tile -> LDS (A frag layout),
//   36x v_wmma_f32_16x16x32_f16 (x@Wx and h@Wh over 9 N-tiles x 2 K-tiles),
//   lane-local gate math, register h update.
// Weight B-fragments are staged to LDS once; the compiler hoists them into
// registers across the 200-step loop (weights-resident-in-VGPRs GEMM chain).
// ---------------------------------------------------------------------------
__global__ __launch_bounds__(128)
void card_gru_kernel(const float* __restrict__ x,
                     const float* __restrict__ Wx,
                     const float* __restrict__ Wh,
                     const float* __restrict__ bi,
                     const float* __restrict__ br,
                     float* __restrict__ cardOut)
{
    __shared__ v16h sWxf[2][9][32];   // [ktile][ntile][lane]  18 KB
    __shared__ v16h sWhf[2][9][32];   // zero-padded K=48..63  18 KB
    __shared__ v16h sXf [4][2][32];   // [mtile][ktile][lane]   8 KB
    __shared__ v16h sHf [4][2][32];   //                        8 KB

    const int tid  = threadIdx.x;
    const int w    = tid >> 5;        // wave / M-tile id
    const int lane = tid & 31;
    const int hiL  = lane >> 4;
    const int c16  = lane & 15;       // C-layout column within n-tile
    const int b0   = blockIdx.x * 64;

    // ---- stage weights once (B-fragment layout: K=e+8*(e>=8)+8*hi, N=lane%16) --
    for (int i = tid; i < F_ * G_; i += 128) {
        const int k  = i / G_;
        const int nc = i - k * G_;
        const _Float16 vx = (_Float16)Wx[i];
        const _Float16 vh = (k < U_) ? (_Float16)Wh[k * G_ + nc] : (_Float16)0.0f;
        const int kt = k >> 5, Kr = k & 31;
        const int hi = (Kr >> 3) & 1, e = (Kr & 7) + ((Kr >> 4) << 3);
        const int n  = nc >> 4;
        const int L  = (nc & 15) + (hi << 4);
        ((_Float16*)&sWxf[kt][n][L])[e] = vx;
        ((_Float16*)&sWhf[kt][n][L])[e] = vh;
    }
    // zero h staging (covers h0=0 and the permanent K=48..63 padding)
    {
        v16h z = {};
        sHf[w][0][lane] = z;
        sHf[w][1][lane] = z;
    }
    __syncthreads();   // weights are written cross-wave

    // ---- per-lane gate biases (column = 16*n + c16, fixed per lane) --------
    float cb[6], bih[3], brh[3];
#pragma unroll
    for (int n = 0; n < 6; ++n) cb[n] = bi[16 * n + c16] + br[16 * n + c16];
#pragma unroll
    for (int m = 0; m < 3; ++m) {
        bih[m] = bi[96 + 16 * m + c16];
        brh[m] = br[96 + 16 * m + c16];
    }

    v8f hreg[3];
#pragma unroll
    for (int m = 0; m < 3; ++m) hreg[m] = (v8f){};

    // ---- sequential recurrence ---------------------------------------------
#pragma unroll 1
    for (int t = 0; t < T_; ++t) {
        // stage x tile, wave-local: 2 threads per row -> rows of own wave
        {
            const int row = tid >> 1;           // 0..63
            const int kt  = tid & 1;            // K chunk of 32
            const int mt  = row >> 4, Lm = row & 15;
            const float* xp = x + (size_t)(b0 + row) * (T_ * F_)
                                + (size_t)t * F_ + kt * 32;
            float xv[32];
            const float4* xp4 = (const float4*)xp;
#pragma unroll
            for (int q = 0; q < 8; ++q) {
                float4 v = xp4[q];
                xv[4 * q + 0] = v.x; xv[4 * q + 1] = v.y;
                xv[4 * q + 2] = v.z; xv[4 * q + 3] = v.w;
            }
            if (t + 1 < T_) __builtin_prefetch(xp + F_, 0, 0);  // global_prefetch_b8
#pragma unroll
            for (int Kr = 0; Kr < 32; ++Kr) {
                int hi, e; amap(Kr, hi, e);
                ((_Float16*)&sXf[mt][kt][Lm + (hi << 4)])[e] = (_Float16)xv[Kr];
            }
        }

        // A fragments (contiguous 32B LDS reads, wave-local data)
        const v16h ax0 = sXf[w][0][lane];
        const v16h ax1 = sXf[w][1][lane];
        const v16h ah0 = sHf[w][0][lane];
        const v16h ah1 = sHf[w][1][lane];

        // z/r gate pre-activations: gx+gh combined (n-tiles 0..5)
        v8f acc[6];
#pragma unroll
        for (int n = 0; n < 6; ++n) {
            v8f c = {};
            c = __builtin_amdgcn_wmma_f32_16x16x32_f16(false, ax0, false, sWxf[0][n][lane], (short)0, c, false, false);
            c = __builtin_amdgcn_wmma_f32_16x16x32_f16(false, ax1, false, sWxf[1][n][lane], (short)0, c, false, false);
            c = __builtin_amdgcn_wmma_f32_16x16x32_f16(false, ah0, false, sWhf[0][n][lane], (short)0, c, false, false);
            c = __builtin_amdgcn_wmma_f32_16x16x32_f16(false, ah1, false, sWhf[1][n][lane], (short)0, c, false, false);
            acc[n] = c;
        }
        // candidate block needs gx and gh separate (r multiplies gh only)
        v8f accx[3], acch[3];
#pragma unroll
        for (int m = 0; m < 3; ++m) {
            const int n = 6 + m;
            v8f cx = {};
            cx = __builtin_amdgcn_wmma_f32_16x16x32_f16(false, ax0, false, sWxf[0][n][lane], (short)0, cx, false, false);
            cx = __builtin_amdgcn_wmma_f32_16x16x32_f16(false, ax1, false, sWxf[1][n][lane], (short)0, cx, false, false);
            accx[m] = cx;
            v8f ch = {};
            ch = __builtin_amdgcn_wmma_f32_16x16x32_f16(false, ah0, false, sWhf[0][n][lane], (short)0, ch, false, false);
            ch = __builtin_amdgcn_wmma_f32_16x16x32_f16(false, ah1, false, sWhf[1][n][lane], (short)0, ch, false, false);
            acch[m] = ch;
        }

        // lane-local gates + h update + re-stage h (f16, A-frag layout)
#pragma unroll
        for (int m = 0; m < 3; ++m) {
#pragma unroll
            for (int r = 0; r < 8; ++r) {
                const float z  = sigf(acc[m][r]     + cb[m]);
                const float rr = sigf(acc[m + 3][r] + cb[m + 3]);
                const float hc = tanh_fast(accx[m][r] + bih[m] + rr * (acch[m][r] + brh[m]));
                const float hn = z * hreg[m][r] + (1.0f - z) * hc;
                hreg[m][r] = hn;
                const int j    = 16 * m + c16;       // hidden column (0..47)
                const int rowl = r + (hiL << 3);     // row within M-tile
                const int kt   = j >> 5, Kr = j & 31;
                int hi, e; amap(Kr, hi, e);
                ((_Float16*)&sHf[w][kt][rowl + (hi << 4)])[e] = (_Float16)hn;
            }
        }
    }

    // ---- write final hidden state ------------------------------------------
#pragma unroll
    for (int m = 0; m < 3; ++m) {
#pragma unroll
        for (int r = 0; r < 8; ++r) {
            const int rowg = b0 + 16 * w + r + (hiL << 3);
            const int col  = 16 * m + c16;
            cardOut[(size_t)rowg * U_ + col] = hreg[m][r];
        }
    }
}

// ---------------------------------------------------------------------------
// Kernel 2: per-step scatter winners (max batch index wins each table slot).
// ---------------------------------------------------------------------------
__global__ __launch_bounds__(256)
void winners_kernel(const float* __restrict__ x, int* __restrict__ win)
{
    __shared__ int wsh[SHARED_];
    const int t = blockIdx.x;
    if (threadIdx.x < SHARED_) wsh[threadIdx.x] = -1;
    __syncthreads();
    for (int b = threadIdx.x; b < B_; b += blockDim.x) {
        const int id = (int)x[(size_t)b * (T_ * F_) + (size_t)t * F_ + 2];
        if (id >= 0 && id < SHARED_) atomicMax(&wsh[id], b);
    }
    __syncthreads();
    if (threadIdx.x < SHARED_) win[t * 16 + threadIdx.x] = wsh[threadIdx.x];
}

// ---------------------------------------------------------------------------
// Kernel 3: sequential 15x48 table evolution for steps 0..T-2 (single block;
// the chain is globally serial, but only 720 state values exist).
// ---------------------------------------------------------------------------
__global__ __launch_bounds__(768)
void table_kernel(const float* __restrict__ x, const float* __restrict__ Wx,
                  const float* __restrict__ Wh, const float* __restrict__ bb,
                  const int* __restrict__ win, float* __restrict__ tableF)
{
    __shared__ float table[SHARED_][U_];
    __shared__ float xr[SHARED_][F_];
    const int tid = threadIdx.x;
    for (int i = tid; i < SHARED_ * U_; i += blockDim.x) table[i / U_][i % U_] = 0.0f;
    __syncthreads();
    const int  k   = tid / U_;
    const int  j   = tid % U_;
    const bool act = (tid < SHARED_ * U_);

    for (int t = 0; t < T_ - 1; ++t) {
        for (int i = tid; i < SHARED_ * F_; i += blockDim.x) {
            const int kk = i / F_, c = i % F_;
            const int b  = win[t * 16 + kk];
            xr[kk][c] = (b >= 0) ? x[(size_t)b * (T_ * F_) + (size_t)t * F_ + c] : 0.0f;
        }
        __syncthreads();
        float hn = 0.0f; bool upd = false;
        if (act && win[t * 16 + k] >= 0) {
            float gz = bb[j], gr = bb[U_ + j], gh = bb[2 * U_ + j];
            for (int c = 0; c < F_; ++c) {
                const float xv = xr[k][c];
                gz += xv * Wx[c * G_ + j];
                gr += xv * Wx[c * G_ + U_ + j];
                gh += xv * Wx[c * G_ + 2 * U_ + j];
            }
            float hz = 0.0f, hr = 0.0f, hh = 0.0f;
            const float hold = table[k][j];
            for (int c = 0; c < U_; ++c) {
                const float hv = table[k][c];
                hz += hv * Wh[c * G_ + j];
                hr += hv * Wh[c * G_ + U_ + j];
                hh += hv * Wh[c * G_ + 2 * U_ + j];
            }
            const float z  = sigf(gz + hz);
            const float r  = sigf(gr + hr);
            const float hc = tanh_fast(gh + r * hh);
            hn = z * hold + (1.0f - z) * hc;
            upd = true;
        }
        __syncthreads();
        if (upd) table[k][j] = hn;
        __syncthreads();
    }
    if (act) tableF[k * U_ + j] = table[k][j];
}

// ---------------------------------------------------------------------------
// Kernel 4: final-step category output for every batch row (parallel).
// ---------------------------------------------------------------------------
__global__ __launch_bounds__(256)
void cat_final_kernel(const float* __restrict__ x, const float* __restrict__ Wx,
                      const float* __restrict__ Wh, const float* __restrict__ bb,
                      const float* __restrict__ tableF, float* __restrict__ catOut)
{
    const int idx = blockIdx.x * blockDim.x + threadIdx.x;
    if (idx >= B_ * U_) return;
    const int b = idx / U_, j = idx % U_;
    const float* xp = x + (size_t)b * (T_ * F_) + (size_t)(T_ - 1) * F_;
    int id = (int)xp[2];
    id = (id < 0) ? 0 : ((id >= SHARED_) ? SHARED_ - 1 : id);
    const float* hp = tableF + id * U_;
    float gz = bb[j], gr = bb[U_ + j], gh = bb[2 * U_ + j];
    for (int c = 0; c < F_; ++c) {
        const float xv = xp[c];
        gz += xv * Wx[c * G_ + j];
        gr += xv * Wx[c * G_ + U_ + j];
        gh += xv * Wx[c * G_ + 2 * U_ + j];
    }
    float hz = 0.0f, hr = 0.0f, hh = 0.0f;
    for (int c = 0; c < U_; ++c) {
        const float hv = hp[c];
        hz += hv * Wh[c * G_ + j];
        hr += hv * Wh[c * G_ + U_ + j];
        hh += hv * Wh[c * G_ + 2 * U_ + j];
    }
    const float z  = sigf(gz + hz);
    const float r  = sigf(gr + hr);
    const float hc = tanh_fast(gh + r * hh);
    catOut[idx] = z * hp[j] + (1.0f - z) * hc;
}

// ---------------------------------------------------------------------------
// Kernel 5: output head  sigmoid([card|cat] @ out_W + out_b)
// ---------------------------------------------------------------------------
__global__ __launch_bounds__(256)
void out_kernel(const float* __restrict__ cardOut, const float* __restrict__ catOut,
                const float* __restrict__ oW, const float* __restrict__ ob,
                float* __restrict__ out)
{
    const int b = blockIdx.x * blockDim.x + threadIdx.x;
    if (b >= B_) return;
    float acc = ob[0];
#pragma unroll
    for (int u = 0; u < U_; ++u) acc += cardOut[b * U_ + u] * oW[u];
#pragma unroll
    for (int u = 0; u < U_; ++u) acc += catOut[b * U_ + u] * oW[U_ + u];
    out[b] = sigf(acc);
}

// ---------------------------------------------------------------------------
extern "C" void kernel_launch(void* const* d_in, const int* in_sizes, int n_in,
                              void* d_out, int out_size, void* d_ws, size_t ws_size,
                              hipStream_t stream)
{
    const float* x       = (const float*)d_in[0];  // [B,T,F]
    const float* card_Wx = (const float*)d_in[1];  // [F,3U]
    const float* card_Wh = (const float*)d_in[2];  // [U,3U]
    const float* card_bi = (const float*)d_in[3];  // [3U]
    const float* card_br = (const float*)d_in[4];  // [3U]
    const float* cat_Wx  = (const float*)d_in[5];  // [F,3U]
    const float* cat_Wh  = (const float*)d_in[6];  // [U,3U]
    const float* cat_b   = (const float*)d_in[7];  // [3U]
    const float* out_W   = (const float*)d_in[8];  // [2U,1]
    const float* out_b   = (const float*)d_in[9];  // [1]
    float* out = (float*)d_out;                    // [B,1]

    // workspace layout
    char*  ws      = (char*)d_ws;
    float* cardOut = (float*)(ws);                                   // B*U f32
    float* catOut  = (float*)(ws + (size_t)B_ * U_ * 4);             // B*U f32
    int*   win     = (int*)  (ws + (size_t)2 * B_ * U_ * 4);         // T*16 i32
    float* tableF  = (float*)(ws + (size_t)2 * B_ * U_ * 4 + T_ * 16 * 4);

    winners_kernel<<<T_, 256, 0, stream>>>(x, win);
    card_gru_kernel<<<B_ / 64, 128, 0, stream>>>(x, card_Wx, card_Wh,
                                                 card_bi, card_br, cardOut);
    table_kernel<<<1, 768, 0, stream>>>(x, cat_Wx, cat_Wh, cat_b, win, tableF);
    cat_final_kernel<<<(B_ * U_ + 255) / 256, 256, 0, stream>>>(
        x, cat_Wx, cat_Wh, cat_b, tableF, catOut);
    out_kernel<<<(B_ + 255) / 256, 256, 0, stream>>>(cardOut, catOut, out_W, out_b, out);
}
